// Spatial_conv_47270410060055
// MI455X (gfx1250) — matrix operand order repaired and checked by reading, not compiled, not run on hardware
//
#include <hip/hip_runtime.h>

typedef __attribute__((ext_vector_type(16))) _Float16 v16h;
typedef __attribute__((ext_vector_type(8)))  _Float16 v8h;
typedef __attribute__((ext_vector_type(4)))  _Float16 v4h;
typedef __attribute__((ext_vector_type(8)))  float    v8f;

#define WB 8
#define WC 128
#define WF 16
#define WN 512

#define WMMA_F16(a, b, c) \
  __builtin_amdgcn_wmma_f32_16x16x32_f16(false, (a), false, (b), (short)0, (c), false, false)

// A-fragment (16x32 f16, MxK) from row-major LDS tile [row][K], pitch in halfs.
// Lane L (r = L&15, hi = L>>4): elems 0..7 -> K = 8*hi + 0..7, elems 8..15 -> K = 8*hi + 16..23.
__device__ __forceinline__ v16h ld_a_frag(const _Float16* base, int pitch) {
  int lane = threadIdx.x & 31;
  int r = lane & 15, hi = lane >> 4;
  const _Float16* p = base + r * pitch + hi * 8;
  v8h lo = *(const v8h*)p;
  v8h hh = *(const v8h*)(p + 16);
  v16h o;
#pragma unroll
  for (int i = 0; i < 8; ++i) { o[i] = lo[i]; o[i + 8] = hh[i]; }
  return o;
}

// B-fragment (32x16 f16, KxN) from K-major LDS tile [col][K], pitch in halfs.
// Lane L (c = L&15, hi = L>>4): elems 0..7 -> K = 8*hi + 0..7, elems 8..15 -> K = 8*hi + 16..23.
__device__ __forceinline__ v16h ld_b_frag(const _Float16* base, int pitch) {
  int lane = threadIdx.x & 31;
  int c = lane & 15, hi = lane >> 4;
  const _Float16* p = base + c * pitch + hi * 8;
  v8h lo = *(const v8h*)p;
  v8h hh = *(const v8h*)(p + 16);
  v16h o;
#pragma unroll
  for (int i = 0; i < 8; ++i) { o[i] = lo[i]; o[i + 8] = hh[i]; }
  return o;
}

// CDNA5 async global->LDS copy (16B per lane), tracked by ASYNCcnt.
__device__ __forceinline__ void async_copy_b128(const _Float16* gaddr, _Float16* laddr) {
  unsigned int l = (unsigned int)(unsigned long long)laddr;  // flat[31:0] == LDS byte offset
  asm volatile("global_load_async_to_lds_b128 %0, %1, off" :: "v"(l), "v"(gaddr) : "memory");
}
__device__ __forceinline__ void wait_async0() {
  asm volatile("s_wait_asynccnt 0" ::: "memory");
}

// ---------------- degree kernel: degInv[b][g][v] = 1 / max(sum_u M[u][v], 1) ----------------
__global__ void spatial_deg(const float* __restrict__ Y, float* __restrict__ degInv) {
  int b = blockIdx.x / 3, g = blockIdx.x % 3;
  int v = threadIdx.x;
  const float* Yp = Y + ((size_t)(b * WF + g)) * WN * WN;
  float cnt = 0.0f;
  for (int u = 0; u < WN; ++u)
    cnt += (Yp[(size_t)u * WN + v] != 0.0f) ? 1.0f : 0.0f;
  degInv[(size_t)(b * 3 + g) * WN + v] = 1.0f / fmaxf(cnt, 1.0f);
}

// ---------------- prep: S = X@(W1+W3)^T, D = X@(W2-W3)^T ----------------
// Emits PXT[b][f][crow][u] f16 with crow 0..127 = P = X.*S, 128..255 = X, and Dp = D + b_dense (f32).
__global__ __launch_bounds__(256) void spatial_prep(
    const float* __restrict__ infos, const float* __restrict__ W_dense,
    const float* __restrict__ b_dense,
    _Float16* __restrict__ PXT, float* __restrict__ Dp) {
  __shared__ _Float16 sm[3 * 128 * 128];  // 96 KB
  _Float16* Ws = sm;            // Wsum[d][c]
  _Float16* Wd = sm + 16384;    // Wdiff[d][c]
  _Float16* Xs = sm + 32768;    // X[n][c]

  int tid = threadIdx.x;
  int w = tid >> 5, lane = tid & 31, hi = lane >> 4, cl = lane & 15;
  int nt = blockIdx.x & 3;
  int f  = (blockIdx.x >> 2) & 15;
  int b  = blockIdx.x >> 6;
  int n0 = nt * 128;

  // Folded weights: float4 reads, packed v4h LDS stores (contiguous in c).
#pragma unroll
  for (int t = 0; t < 16; ++t) {
    int idx4 = tid + t * 256;            // 4096 chunks of 4
    int d = idx4 >> 5, c4 = (idx4 & 31) << 2;
    float4 a1 = *(const float4*)(W_dense + d * 384 + c4);
    float4 a2 = *(const float4*)(W_dense + d * 384 + 128 + c4);
    float4 a3 = *(const float4*)(W_dense + d * 384 + 256 + c4);
    v4h ws, wd;
    ws[0] = (_Float16)(a1.x + a3.x); ws[1] = (_Float16)(a1.y + a3.y);
    ws[2] = (_Float16)(a1.z + a3.z); ws[3] = (_Float16)(a1.w + a3.w);
    wd[0] = (_Float16)(a2.x - a3.x); wd[1] = (_Float16)(a2.y - a3.y);
    wd[2] = (_Float16)(a2.z - a3.z); wd[3] = (_Float16)(a2.w - a3.w);
    *(v4h*)(Ws + d * 128 + c4) = ws;
    *(v4h*)(Wd + d * 128 + c4) = wd;
  }
  // X tile: coalesced float4 over n, transpose-scatter into A-friendly [n][c].
#pragma unroll
  for (int t = 0; t < 16; ++t) {
    int idx4 = tid + t * 256;
    int c = idx4 >> 5, n4 = (idx4 & 31) << 2;
    float4 x = *(const float4*)(infos + (((size_t)b * WC + c) * WF + f) * WN + n0 + n4);
    Xs[(n4 + 0) * 128 + c] = (_Float16)x.x;
    Xs[(n4 + 1) * 128 + c] = (_Float16)x.y;
    Xs[(n4 + 2) * 128 + c] = (_Float16)x.z;
    Xs[(n4 + 3) * 128 + c] = (_Float16)x.w;
  }
  __syncthreads();

  v8f accS[8] = {}, accD[8] = {};
#pragma unroll
  for (int k0 = 0; k0 < 128; k0 += 32) {
    v16h a = ld_a_frag(Xs + (w * 16) * 128 + k0, 128);
#pragma unroll
    for (int ct = 0; ct < 8; ++ct) {
      v16h bs = ld_b_frag(Ws + (ct * 16) * 128 + k0, 128);
      accS[ct] = WMMA_F16(a, bs, accS[ct]);
      v16h bd = ld_b_frag(Wd + (ct * 16) * 128 + k0, 128);
      accD[ct] = WMMA_F16(a, bd, accD[ct]);
    }
  }

  _Float16* PXp = PXT + (size_t)(b * WF + f) * 256 * WN;
  float* DpP = Dp + (size_t)(b * WF + f) * WN * WC;
  size_t ub = (size_t)(n0 + w * 16 + 8 * hi);   // 8 consecutive u per lane
#pragma unroll
  for (int ct = 0; ct < 8; ++ct) {
    int d = ct * 16 + cl;
    float bd = b_dense[d];
    v8h pv, xv;
#pragma unroll
    for (int r = 0; r < 8; ++r) {
      int n = w * 16 + 8 * hi + r;
      float x = (float)Xs[n * 128 + d];
      xv[r] = (_Float16)x;
      pv[r] = (_Float16)(x * accS[ct][r]);
      DpP[(size_t)(n0 + n) * WC + d] = accD[ct][r] + bd;
    }
    *(v8h*)(PXp + (size_t)d * WN + ub)         = pv;  // P rows (b128 store)
    *(v8h*)(PXp + (size_t)(128 + d) * WN + ub) = xv;  // X rows (b128 store)
  }
}

// ---------------- main: [A1|A2] = M^T @ [P|X]; agg = (A1 + A2*Dp)*degInv; out = relu(agg@Wn^T + bn) ----------------
__global__ __launch_bounds__(256) void spatial_main(
    const float* __restrict__ Y, const _Float16* __restrict__ PXT,
    const float* __restrict__ Dp, const float* __restrict__ degInv,
    const float* __restrict__ W_node, const float* __restrict__ b_node,
    float* __restrict__ out) {
  __shared__ __align__(16) char smem[98304];
  // Stage buffers (ping-pong): buf q at +q*24576: As[128v][32u] (8KB) + PXs[256c][32u] (16KB)
  _Float16* WnL  = (_Float16*)(smem + 49152);   // [128 o][128 c] 32 KB (staged once)
  _Float16* aggL = (_Float16*)smem;             // [128 v][128 c] 32 KB (phase B)
  float*    hT   = (float*)smem;                // [128 o][128 v] 64 KB (phase C)

  int tid = threadIdx.x;
  int w = tid >> 5, lane = tid & 31, hi = lane >> 4, cl = lane & 15;
  int vt = blockIdx.x & 3;
  int f  = (blockIdx.x >> 2) & 15;
  int b  = blockIdx.x >> 6;
  int g  = (f == 0) ? 0 : ((f <= 11) ? 1 : 2);
  int v0 = vt * 128;

  for (int idx = tid; idx < 128 * 128; idx += 256)   // stage W_node (f16)
    WnL[idx] = (_Float16)W_node[idx];

  const float* Yp = Y + ((size_t)(b * WF + g)) * WN * WN;
  const _Float16* PXp = PXT + (size_t)(b * WF + f) * 256 * WN;

  // Stage one 32-u-wide K tile into buffer q: mask (compare+convert) + async row copy of PXT.
  auto stage = [&](int q, int k0) {
    _Float16* As  = (_Float16*)(smem + q * 24576);
    _Float16* PXs = (_Float16*)(smem + q * 24576 + 8192);
#pragma unroll
    for (int t = 0; t < 4; ++t) {                    // 1024 float4 chunks of Y
      int idx4 = tid + t * 256;
      int ul = idx4 >> 5, vl = (idx4 & 31) << 2;
      float4 y = *(const float4*)(Yp + (size_t)(k0 + ul) * WN + v0 + vl);
      As[(vl + 0) * 32 + ul] = (y.x != 0.0f) ? (_Float16)1.0f : (_Float16)0.0f;
      As[(vl + 1) * 32 + ul] = (y.y != 0.0f) ? (_Float16)1.0f : (_Float16)0.0f;
      As[(vl + 2) * 32 + ul] = (y.z != 0.0f) ? (_Float16)1.0f : (_Float16)0.0f;
      As[(vl + 3) * 32 + ul] = (y.w != 0.0f) ? (_Float16)1.0f : (_Float16)0.0f;
    }
#pragma unroll
    for (int t = 0; t < 4; ++t) {                    // 1024 16B chunks of PXT rows
      int q2 = tid + t * 256;
      int c = q2 >> 2, seg = (q2 & 3) << 3;          // seg in halfs
      async_copy_b128(PXp + (size_t)c * WN + k0 + seg, PXs + c * 32 + seg);
    }
  };

  v8f acc[16] = {};  // tiles: cols 0..127 -> A1, 128..255 -> A2
  stage(0, 0);
  wait_async0();
  __syncthreads();
  for (int i = 0; i < 16; ++i) {
    _Float16* As  = (_Float16*)(smem + (i & 1) * 24576);
    _Float16* PXs = (_Float16*)(smem + (i & 1) * 24576 + 8192);
    v16h a = ld_a_frag(As + (w * 16) * 32, 32);      // reused across 16 tiles
#pragma unroll
    for (int ct = 0; ct < 16; ++ct) {
      v16h bf = ld_b_frag(PXs + (ct * 16) * 32, 32);
      acc[ct] = WMMA_F16(a, bf, acc[ct]);
    }
    if (i + 1 < 16) stage((i + 1) & 1, (i + 1) * 32); // overlap staging with WMMA drain
    wait_async0();
    __syncthreads();
  }

  // agg = (A1 + A2 * Dp) * degInv  -> LDS (f16), A-friendly [v][c]
  const float* dinv = degInv + ((size_t)(b * 3 + g)) * WN + v0;
  const float* DpP  = Dp + (size_t)(b * WF + f) * WN * WC + (size_t)v0 * WC;
#pragma unroll
  for (int ct = 0; ct < 8; ++ct) {
#pragma unroll
    for (int r = 0; r < 8; ++r) {
      int vl = w * 16 + r + 8 * hi;
      int c  = ct * 16 + cl;
      float gv = (acc[ct][r] + acc[ct + 8][r] * DpP[(size_t)vl * WC + c]) * dinv[vl];
      aggL[vl * 128 + c] = (_Float16)gv;
    }
  }
  __syncthreads();

  // Node GEMM: h = agg @ W_node^T
  v8f acch[8] = {};
#pragma unroll
  for (int k0 = 0; k0 < 128; k0 += 32) {
    v16h a = ld_a_frag(aggL + (w * 16) * 128 + k0, 128);
#pragma unroll
    for (int ot = 0; ot < 8; ++ot) {
      v16h bf = ld_b_frag(WnL + (ot * 16) * 128 + k0, 128);
      acch[ot] = WMMA_F16(a, bf, acch[ot]);
    }
  }
  __syncthreads();

  // relu(+bias), transpose in LDS, write [B,C,F,N] coalesced
#pragma unroll
  for (int ot = 0; ot < 8; ++ot) {
#pragma unroll
    for (int r = 0; r < 8; ++r) {
      int vl = w * 16 + r + 8 * hi;
      int o  = ot * 16 + cl;
      hT[o * 128 + vl] = fmaxf(acch[ot][r] + b_node[o], 0.0f);
    }
  }
  __syncthreads();
  for (int idx = tid; idx < 128 * 32; idx += 256) {
    int o = idx >> 5, vq = (idx & 31) << 2;
    float4 val = *(const float4*)(hT + o * 128 + vq);
    *(float4*)(out + ((size_t)(b * WC + o) * WF + f) * WN + v0 + vq) = val;
  }
}

extern "C" void kernel_launch(void* const* d_in, const int* in_sizes, int n_in,
                              void* d_out, int out_size, void* d_ws, size_t ws_size,
                              hipStream_t stream) {
  const float* Y       = (const float*)d_in[0];
  const float* infos   = (const float*)d_in[1];
  const float* W_dense = (const float*)d_in[2];
  const float* b_dense = (const float*)d_in[3];
  const float* W_node  = (const float*)d_in[4];
  const float* b_node  = (const float*)d_in[5];
  float* out = (float*)d_out;

  char* ws = (char*)d_ws;
  _Float16* PXT = (_Float16*)ws;                      // [B*F][256 crow][512 u] f16 = 32 MB
  float*    Dp  = (float*)(ws + (size_t)33554432);    // [B*F][512 u][128 d] f32 = 32 MB
  float*    dgi = (float*)(ws + (size_t)67108864);    // [B][3][512] f32 = 48 KB

  spatial_deg <<<WB * 3, WN, 0, stream>>>(Y, dgi);
  spatial_prep<<<WB * WF * 4, 256, 0, stream>>>(infos, W_dense, b_dense, PXT, Dp);
  spatial_main<<<WB * WF * 4, 256, 0, stream>>>(Y, PXT, Dp, dgi, W_node, b_node, out);
}
